// ISTFT_Process_41162966565052
// MI455X (gfx1250) — matrix operand order repaired and checked by reading, not compiled, not run on hardware
//
#include <hip/hip_runtime.h>

// ---- problem constants -------------------------------------------------
#define BATCH   8
#define T_FR    345          // frames
#define TP      384          // padded frames (3 x 128 row-blocks)
#define NBINS   2049         // N_FFT/2+1
#define C_IN    4098         // 2*NBINS (GEMM K dimension)
#define CP      4128         // padded K (129 x 32)
#define NK      4096         // N_FFT (GEMM N dimension)
#define HOPS    1024
#define HALF    2048
#define OUTLEN  352800
#define EPSF    1.19209290e-7f

typedef __attribute__((ext_vector_type(16))) __bf16 v16bf;
typedef __attribute__((ext_vector_type(8)))  float  v8f;

// ---- kernel 1: polar round-trip + transpose + bf16 pack ----------------
__global__ __launch_bounds__(256) void pack_a(const float* __restrict__ real,
                                              const float* __restrict__ imag,
                                              __bf16* __restrict__ A) {
  size_t idx = (size_t)blockIdx.x * 256 + threadIdx.x;   // over BATCH*TP*CP
  int    c   = (int)(idx % CP);
  size_t bt  = idx / CP;
  int    t   = (int)(bt % TP);
  int    b   = (int)(bt / TP);
  float val = 0.f;
  if (t < T_FR && c < C_IN) {
    int cc = (c < NBINS) ? c : (c - NBINS);
    size_t off = ((size_t)b * NBINS + cc) * T_FR + t;
    float r  = real[off];
    float im = imag[off];
    float rp  = r + EPSF;
    float mag = sqrtf(r * r + im * im);
    float hyp = sqrtf(rp * rp + im * im);
    float inv = (hyp > 0.f) ? (mag / hyp) : 0.f;
    val = (c < NBINS) ? (rp * inv) : (im * inv);
  }
  A[idx] = (__bf16)val;
}

// ---- kernel 2: basis transpose + bf16 pack -----------------------------
__global__ __launch_bounds__(256) void pack_b(const float* __restrict__ basis,
                                              __bf16* __restrict__ Bt) {
  size_t idx = (size_t)blockIdx.x * 256 + threadIdx.x;   // over NK*CP
  int    c   = (int)(idx % CP);
  size_t n   = idx / CP;
  float val = (c < C_IN) ? basis[(size_t)c * NK + n] : 0.f;
  Bt[idx] = (__bf16)val;
}

// ---- async global->LDS tile copy (CDNA5, ASYNCcnt-tracked) -------------
// Copies one 128x32 bf16 tile of A and of B into one LDS buffer.
__device__ __forceinline__ void async_tile(unsigned ldsA, unsigned ldsB,
                                           unsigned voff,
                                           const __bf16* Ab, const __bf16* Bb) {
  unsigned long long sa = (unsigned long long)Ab;
  unsigned long long sb = (unsigned long long)Bb;
  unsigned voff1 = voff + 64u * CP * 2u;     // +64 rows in global (bytes)
  asm volatile("global_load_async_to_lds_b128 %0, %1, %2"
               :: "v"(ldsA), "v"(voff), "s"(sa) : "memory");
  asm volatile("global_load_async_to_lds_b128 %0, %1, %2"
               :: "v"(ldsA + 4096u), "v"(voff1), "s"(sa) : "memory");
  asm volatile("global_load_async_to_lds_b128 %0, %1, %2"
               :: "v"(ldsB), "v"(voff), "s"(sb) : "memory");
  asm volatile("global_load_async_to_lds_b128 %0, %1, %2"
               :: "v"(ldsB + 4096u), "v"(voff1), "s"(sb) : "memory");
}

#define ASYNC_WAIT() asm volatile("s_wait_asynccnt 0x0" ::: "memory")

// ---- fragment load + 8x WMMA for one 32-K LDS tile ---------------------
__device__ __forceinline__ void frag_mma(const __bf16 (*__restrict__ Asb)[32],
                                         const __bf16 (*__restrict__ Bsb)[32],
                                         int wrow, int wcol, int m, int h,
                                         v8f acc[4][2]) {
  // A fragments: 16x32 bf16; lane half h owns K offset +8 per pair group.
  v16bf afrag[4];
#pragma unroll
  for (int mt = 0; mt < 4; ++mt) {
    const __bf16* ap = &Asb[wrow * 64 + mt * 16 + m][0];
    unsigned int* dst = (unsigned int*)&afrag[mt];
#pragma unroll
    for (int p = 0; p < 8; ++p) {
      int kk = ((p < 4) ? (2 * p) : (2 * p + 8)) + 8 * h;
      dst[p] = *(const unsigned int*)(ap + kk);
    }
  }
  // B fragments: 32x16 bf16; lane half h owns K [8h, 8h+8) and [16+8h, 24+8h).
  v16bf bfrag[2];
#pragma unroll
  for (int nt = 0; nt < 2; ++nt) {
    const __bf16* bp = &Bsb[wcol * 32 + nt * 16 + m][8 * h];
    ((uint4*)&bfrag[nt])[0] = *(const uint4*)(bp);
    ((uint4*)&bfrag[nt])[1] = *(const uint4*)(bp + 16);
  }
#pragma unroll
  for (int mt = 0; mt < 4; ++mt)
#pragma unroll
    for (int nt = 0; nt < 2; ++nt)
      acc[mt][nt] = __builtin_amdgcn_wmma_f32_16x16x32_bf16(
          false, afrag[mt], false, bfrag[nt],
          (short)0, acc[mt][nt], false, false);
}

// ---- kernel 3: WMMA GEMM  frames[b][t][k] = sum_c A[b][t][c]*Bt[k][c] --
// Block tile 128(M) x 128(N), K-step 32, 8 waves (wave32).
// Double-buffered LDS fed by async-to-LDS loads; manual 2-stage software
// pipeline with compile-time buffer indices (129 tiles = 1 + 64*2).
__global__ __launch_bounds__(256) void gemm_wmma(const __bf16* __restrict__ A,
                                                 const __bf16* __restrict__ Bt,
                                                 float* __restrict__ frames) {
  __shared__ __bf16 As[2][128][32];   // 8 KB per buffer
  __shared__ __bf16 Bs[2][128][32];

  const int tid  = threadIdx.x;
  const int lane = tid & 31;
  const int wave = tid >> 5;
  const int wrow = wave & 1;       // M half    (64 rows)
  const int wcol = wave >> 1;      // N quarter (32 cols)
  const int nblk = blockIdx.x;     // 0..31
  const int mblk = blockIdx.y;     // 0..2
  const int b    = blockIdx.z;     // 0..7

  const __bf16* Ab = A  + ((size_t)b * TP + (size_t)mblk * 128) * CP;
  const __bf16* Bb = Bt + (size_t)nblk * 128 * CP;

  const int ldr = tid >> 2;          // 0..63 : tile row this thread copies
  const int ldc = (tid & 3) * 8;     // 0,8,16,24 : 16B chunk
  const int m   = lane & 15;
  const int h   = lane >> 4;

  unsigned voff      = (unsigned)((ldr * CP + ldc) * 2);   // bytes; +64/step
  unsigned ldsA_base = (unsigned)(uintptr_t)&As[0][ldr][ldc];
  unsigned ldsB_base = (unsigned)(uintptr_t)&Bs[0][ldr][ldc];
  const unsigned SEL = 8192u;        // buffer stride (128*32*2 bytes)

  v8f acc[4][2] = {};

  // prologue: tile 0 -> buf 0
  async_tile(ldsA_base, ldsB_base, voff, Ab, Bb);
  voff += 64u;
  ASYNC_WAIT();
  __syncthreads();

  // steady state: tiles 1..128 prefetched, tiles 0..127 computed
  for (int i = 0; i < 64; ++i) {
    async_tile(ldsA_base + SEL, ldsB_base + SEL, voff, Ab, Bb);   // -> buf 1
    voff += 64u;
    frag_mma(As[0], Bs[0], wrow, wcol, m, h, acc);
    ASYNC_WAIT();
    __syncthreads();

    async_tile(ldsA_base, ldsB_base, voff, Ab, Bb);               // -> buf 0
    voff += 64u;
    frag_mma(As[1], Bs[1], wrow, wcol, m, h, acc);
    ASYNC_WAIT();
    __syncthreads();
  }
  // epilogue: tile 128 in buf 0
  frag_mma(As[0], Bs[0], wrow, wcol, m, h, acc);

  // store: D layout -> rows M = 8h + r, cols N = lane&15
  const int colbase = nblk * 128 + wcol * 32 + m;
#pragma unroll
  for (int mt = 0; mt < 4; ++mt) {
    int rowbase = mblk * 128 + wrow * 64 + mt * 16 + 8 * h;
#pragma unroll
    for (int nt = 0; nt < 2; ++nt) {
      int col = colbase + nt * 16;
#pragma unroll
      for (int r = 0; r < 8; ++r) {
        int row = rowbase + r;
        if (row < T_FR)
          frames[((size_t)b * T_FR + row) * NK + col] = acc[mt][nt][r];
      }
    }
  }
}

// ---- kernel 4: overlap-add (gather, deterministic) + normalize ---------
__global__ __launch_bounds__(256) void ola(const float* __restrict__ frames,
                                           const float* __restrict__ wsi,
                                           float* __restrict__ out) {
  size_t idx = (size_t)blockIdx.x * 256 + threadIdx.x;   // over BATCH*OUTLEN
  int j = (int)(idx % OUTLEN);
  int b = (int)(idx / OUTLEN);
  int n = j + HALF;
  int tlo = (n - (NK - HOPS)) >> 10;
  if (tlo < 0) tlo = 0;
  int thi = n >> 10;
  if (thi > T_FR - 1) thi = T_FR - 1;
  float s = 0.f;
  for (int t = tlo; t <= thi; ++t)
    s += frames[((size_t)b * T_FR + t) * NK + (n - (t << 10))];
  out[idx] = s * wsi[n];
}

// ---- launcher ----------------------------------------------------------
extern "C" void kernel_launch(void* const* d_in, const int* in_sizes, int n_in,
                              void* d_out, int out_size, void* d_ws, size_t ws_size,
                              hipStream_t stream) {
  const float* real  = (const float*)d_in[0];
  const float* imag  = (const float*)d_in[1];
  const float* basis = (const float*)d_in[2];   // [4098][4096]
  const float* wsi   = (const float*)d_in[3];   // [356352]
  (void)in_sizes; (void)n_in; (void)out_size; (void)ws_size;

  __bf16* A      = (__bf16*)d_ws;                          // [8][384][4128]
  __bf16* Bt     = A + (size_t)BATCH * TP * CP;            // [4096][4128]
  float*  frames = (float*)(Bt + (size_t)NK * CP);         // [8][345][4096]

  pack_a<<<(BATCH * TP * CP) / 256, 256, 0, stream>>>(real, imag, A);
  pack_b<<<(NK * CP) / 256, 256, 0, stream>>>(basis, Bt);
  gemm_wmma<<<dim3(NK / 128, TP / 128, BATCH), 256, 0, stream>>>(A, Bt, frames);
  ola<<<(BATCH * OUTLEN) / 256, 256, 0, stream>>>(frames, wsi, (float*)d_out);
}
